// GEDModelHomologyGAT_77575699300933
// MI455X (gfx1250) — matrix operand order repaired and checked by reading, not compiled, not run on hardware
//
#include <hip/hip_runtime.h>
#include <hip/hip_bf16.h>
#include <math.h>

#define INF_COST 10000.0f
#define MROWS    4097            // 2048 x_s + 2048 x_t + 1 virtual
#define DFEAT    512
#define NB       16
#define NPAD     129             // N+1 (virtual row appended)
#define EDSZ     (NPAD * NPAD)   // 16641
#define NN       256             // LSAP matrix size (2N)

typedef __attribute__((ext_vector_type(16))) __bf16 v16bf;
typedef __attribute__((ext_vector_type(8)))  __bf16 v8bf;
typedef __attribute__((ext_vector_type(8)))  float  v8f;

// fp32 -> bf16, round-to-nearest-even (bit-level)
__device__ __forceinline__ __bf16 f2bf(float f) {
    union { float f; unsigned u; } cv; cv.f = f;
    unsigned r = cv.u + 0x7FFFu + ((cv.u >> 16) & 1u);
    union { unsigned short s; __bf16 b; } cv2;
    cv2.s = (unsigned short)(r >> 16);
    return cv2.b;
}
// two packed bf16 inside a u32 -> floats
__device__ __forceinline__ float bf_lo(unsigned u) { union { unsigned u; float f; } c; c.u = u << 16;         return c.f; }
__device__ __forceinline__ float bf_hi(unsigned u) { union { unsigned u; float f; } c; c.u = u & 0xFFFF0000u; return c.f; }

__device__ __forceinline__ v16bf cat16(v8bf lo, v8bf hi) {
    return __builtin_shufflevector(lo, hi, 0,1,2,3,4,5,6,7,8,9,10,11,12,13,14,15);
}

// ---------------------------------------------------------------------------
// One-shot fp32 -> bf16 conversion (8 elements / thread, b128 in, b128 out)
// ---------------------------------------------------------------------------
__global__ __launch_bounds__(256)
void cvt_f32_bf16(const float* __restrict__ in, __bf16* __restrict__ out, int n)
{
    const int i = (blockIdx.x * 256 + threadIdx.x) * 8;
    if (i < n) {
        float4 a = *(const float4*)(in + i);
        float4 b = *(const float4*)(in + i + 4);
        v8bf r;
        r[0] = f2bf(a.x); r[1] = f2bf(a.y); r[2] = f2bf(a.z); r[3] = f2bf(a.w);
        r[4] = f2bf(b.x); r[5] = f2bf(b.y); r[6] = f2bf(b.z); r[7] = f2bf(b.w);
        *(v8bf*)(out + i) = r;
    }
}

// ---------------------------------------------------------------------------
// Y[M x 512] = relu( X @ W^T + bias ) in bf16->bf16, f32 accumulate.
// One wave per 16x16 tile; inner loop = 4x global_load_b128 + 1 wmma.
// OOB A-rows read a clamped valid row; they only feed masked D-rows.
// ---------------------------------------------------------------------------
__global__ __launch_bounds__(32)
void gemm_relu_wmma(const __bf16* __restrict__ X, const __bf16* __restrict__ W,
                    const float* __restrict__ bias, __bf16* __restrict__ Y, int M)
{
    const int tm   = blockIdx.x;
    const int tn   = blockIdx.y;
    const int lane = threadIdx.x;
    const int half = lane >> 4;
    const int l16  = lane & 15;

    const int arow   = tm * 16 + l16;
    const int arow_c = arow < MROWS ? arow : MROWS - 1;
    const __bf16* Xrow = X + (size_t)arow_c * DFEAT;
    const __bf16* Wrow = W + (size_t)(tn * 16 + l16) * DFEAT;   // B = W^T column

    v8f acc = {};
#pragma unroll 8
    for (int kk = 0; kk < DFEAT; kk += 32) {
        // A 16x32: lanes 0-15 K = kk+0..7 & kk+16..23; lanes 16-31: +8
        const int ka = kk + half * 8;
        v16bf a = cat16(*(const v8bf*)(Xrow + ka),
                        *(const v8bf*)(Xrow + ka + 16));
        // B 32x16: lane column l16; lanes 0-15 K = kk+0..7 & kk+16..23; lanes 16-31: +8
        const int kb = kk + half * 8;
        v16bf b = cat16(*(const v8bf*)(Wrow + kb),
                        *(const v8bf*)(Wrow + kb + 16));
        acc = __builtin_amdgcn_wmma_f32_16x16x32_bf16(
                  false, a, false, b, (short)0, acc, false, false);
    }

    // D layout: VGPR r -> M = r (lanes 0-15) / 8+r (lanes 16-31); N = l16
    const int col = tn * 16 + l16;
    const float bc = bias[col];
#pragma unroll
    for (int r = 0; r < 8; ++r) {
        int row = tm * 16 + (half ? 8 + r : r);
        if (row < M) {
            float yv = acc[r] + bc;
            Y[(size_t)row * DFEAT + col] = f2bf(yv > 0.0f ? yv : 0.0f);
        }
    }
}

// ---------------------------------------------------------------------------
// Per-row sum of squares of bf16 embeddings (1 row / 64-thread block:
// exactly one uint4 = 8 bf16 per thread)
// ---------------------------------------------------------------------------
__global__ __launch_bounds__(64)
void row_sumsq(const __bf16* __restrict__ E, float* __restrict__ ss)
{
    const int row = blockIdx.x;
    __shared__ float red[64];
    const uint4 v = ((const uint4*)(E + (size_t)row * DFEAT))[threadIdx.x];
    float s = 0.0f;
    s += bf_lo(v.x) * bf_lo(v.x) + bf_hi(v.x) * bf_hi(v.x);
    s += bf_lo(v.y) * bf_lo(v.y) + bf_hi(v.y) * bf_hi(v.y);
    s += bf_lo(v.z) * bf_lo(v.z) + bf_hi(v.z) * bf_hi(v.z);
    s += bf_lo(v.w) * bf_lo(v.w) + bf_hi(v.w) * bf_hi(v.w);
    red[threadIdx.x] = s; __syncthreads();
    for (int off = 32; off; off >>= 1) {
        if (threadIdx.x < off) red[threadIdx.x] += red[threadIdx.x + off];
        __syncthreads();
    }
    if (threadIdx.x == 0) ss[row] = red[0];
}

__device__ __forceinline__ int es_index(int b, int i) { return (i < 128) ? (b * 128 + i) : 4096; }
__device__ __forceinline__ int et_index(int b, int i) { return (i < 128) ? (2048 + b * 128 + i) : 4096; }

// ---------------------------------------------------------------------------
// d[b,i,j] = || es_i - et_j || via WMMA Gram matrix:  d^2 = ssi + ssj - 2*G
// ---------------------------------------------------------------------------
__global__ __launch_bounds__(32)
void cdist_wmma(const __bf16* __restrict__ E, const float* __restrict__ ss,
                float* __restrict__ edit)
{
    const int ti = blockIdx.x, tj = blockIdx.y, b = blockIdx.z;
    const int lane = threadIdx.x, half = lane >> 4, l16 = lane & 15;

    const int ia = ti * 16 + l16;
    const int jb = tj * 16 + l16;
    const int ia_c = ia < NPAD ? ia : 128;
    const int jb_c = jb < NPAD ? jb : 128;
    const __bf16* Arow = E + (size_t)es_index(b, ia_c) * DFEAT;
    const __bf16* Brow = E + (size_t)et_index(b, jb_c) * DFEAT;

    v8f acc = {};
#pragma unroll 8
    for (int kk = 0; kk < DFEAT; kk += 32) {
        const int ka = kk + half * 8;
        v16bf a = cat16(*(const v8bf*)(Arow + ka),
                        *(const v8bf*)(Arow + ka + 16));
        const int kb = kk + half * 8;
        v16bf bf = cat16(*(const v8bf*)(Brow + kb),
                         *(const v8bf*)(Brow + kb + 16));
        acc = __builtin_amdgcn_wmma_f32_16x16x32_bf16(
                  false, a, false, bf, (short)0, acc, false, false);
    }

    const int j = tj * 16 + l16;
    const float ssj = (j < NPAD) ? ss[et_index(b, j)] : 0.0f;
#pragma unroll
    for (int r = 0; r < 8; ++r) {
        int i = ti * 16 + (half ? 8 + r : r);
        if (i < NPAD && j < NPAD) {
            float ssi = ss[es_index(b, i)];
            float sq  = ssi + ssj - 2.0f * acc[r];
            sq = sq > 0.0f ? sq : 0.0f;
            edit[(size_t)b * EDSZ + i * NPAD + j] = sq > 0.0f ? sqrtf(sq) : 0.0f;
        }
    }
}

// ---------------------------------------------------------------------------
// Per-batch sum, then in-place scale to N*N / sum
// ---------------------------------------------------------------------------
__global__ __launch_bounds__(256)
void batch_sum(const float* __restrict__ edit, float* __restrict__ sums)
{
    const int b = blockIdx.x;
    __shared__ float red[256];
    float s = 0.0f;
    for (int t = threadIdx.x; t < EDSZ; t += 256) s += edit[(size_t)b * EDSZ + t];
    red[threadIdx.x] = s; __syncthreads();
    for (int off = 128; off; off >>= 1) {
        if (threadIdx.x < off) red[threadIdx.x] += red[threadIdx.x + off];
        __syncthreads();
    }
    if (threadIdx.x == 0) sums[b] = red[0];
}

__global__ __launch_bounds__(256)
void batch_scale(float* __restrict__ edit, const float* __restrict__ sums)
{
    const int b = blockIdx.y;
    const int t = blockIdx.x * 256 + threadIdx.x;
    if (t < EDSZ) edit[(size_t)b * EDSZ + t] *= (128.0f * 128.0f) / sums[b];
}

// ---------------------------------------------------------------------------
// LSAP cost C(i0,j) (1-based), synthesized from the edit block on the fly
// ---------------------------------------------------------------------------
__device__ __forceinline__ float costC(const float* __restrict__ ec, int i0, int j)
{
    const int r = i0 - 1, c = j - 1;
    if (r < 128) {
        if (c < 128) return ec[r * NPAD + c];
        return (c - 128 == r) ? ec[r * NPAD + 128] : INF_COST;
    } else {
        if (c < 128) return (r - 128 == c) ? ec[128 * NPAD + c] : INF_COST;
        return 0.0f;
    }
}

// ---------------------------------------------------------------------------
// Hungarian (JV) per batch: 1 workgroup / batch, thread t owns column t+1.
// ---------------------------------------------------------------------------
__global__ __launch_bounds__(256)
void hungarian(const float* __restrict__ edit, float* __restrict__ Aout,
               float* __restrict__ geds)
{
    const int b = blockIdx.x, tid = threadIdx.x;
    const float* ec = edit + (size_t)b * EDSZ;
    float* Ab = Aout + (size_t)b * EDSZ;

    __shared__ float u[NN + 1], v[NN + 1], minv[NN + 1];
    __shared__ int   p[NN + 1], way[NN + 1];
    __shared__ unsigned char used[NN + 1];
    __shared__ float rmin[256];
    __shared__ int   ridx[256];
    __shared__ int   s_j0;
    __shared__ float s_delta;

    for (int t = tid; t < EDSZ; t += 256) Ab[t] = 0.0f;
    for (int j = tid; j <= NN; j += 256) { u[j] = 0.0f; v[j] = 0.0f; p[j] = 0; way[j] = 0; }
    __syncthreads();

    for (int i = 1; i <= NN; ++i) {
        if (tid == 0) { p[0] = i; s_j0 = 0; }
        for (int j = tid; j <= NN; j += 256) { minv[j] = 3.0e38f; used[j] = 0; }
        __syncthreads();

        while (true) {
            if (tid == 0) used[s_j0] = 1;
            __syncthreads();
            const int j0 = s_j0;
            const int i0 = p[j0];

            const int j = tid + 1;
            float mv = 3.0e38f;
            if (!used[j]) {
                float cur = costC(ec, i0, j) - u[i0] - v[j];
                if (cur < minv[j]) { minv[j] = cur; way[j] = j0; }
                mv = minv[j];
            }
            rmin[tid] = mv; ridx[tid] = j;
            __syncthreads();
            for (int off = 128; off; off >>= 1) {                  // argmin, ties keep the lower slot
                if (tid < off && rmin[tid + off] < rmin[tid]) {
                    rmin[tid] = rmin[tid + off]; ridx[tid] = ridx[tid + off];
                }
                __syncthreads();
            }
            if (tid == 0) { s_delta = rmin[0]; s_j0 = ridx[0]; }
            __syncthreads();

            const float delta = s_delta;
            for (int jj = tid; jj <= NN; jj += 256) {
                if (used[jj]) { u[p[jj]] += delta; v[jj] -= delta; }
                else          { minv[jj] -= delta; }
            }
            __syncthreads();
            if (p[s_j0] == 0) break;
        }

        if (tid == 0) {
            int j0 = s_j0;
            while (j0) { int j1 = way[j0]; p[j0] = p[j1]; j0 = j1; }
        }
        __syncthreads();
    }

    for (int j = tid + 1; j <= NN; j += 256) {
        int row = p[j] - 1, col = j - 1;
        int r = row < 128 ? row : 128;
        int c = col < 128 ? col : 128;
        Ab[r * NPAD + c] = 1.0f;
    }
    __syncthreads();
    if (tid == 0) Ab[128 * NPAD + 128] = 0.0f;
    __syncthreads();

    float s = 0.0f;
    for (int t = tid; t < EDSZ; t += 256) s += Ab[t] * ec[t];
    rmin[tid] = s; __syncthreads();
    for (int off = 128; off; off >>= 1) {
        if (tid < off) rmin[tid] += rmin[tid + off];
        __syncthreads();
    }
    if (tid == 0) geds[b] = rmin[0] / (float)NN;
}

// ---------------------------------------------------------------------------
extern "C" void kernel_launch(void* const* d_in, const int* in_sizes, int n_in,
                              void* d_out, int out_size, void* d_ws, size_t ws_size,
                              hipStream_t stream)
{
    const float* x_s  = (const float*)d_in[0];
    const float* x_t  = (const float*)d_in[1];
    const float* W1   = (const float*)d_in[2];
    const float* b1   = (const float*)d_in[3];
    const float* W2   = (const float*)d_in[4];
    const float* b2   = (const float*)d_in[5];
    const float* virt = (const float*)d_in[6];
    (void)in_sizes; (void)n_in; (void)out_size; (void)ws_size;

    float* out   = (float*)d_out;
    float* A_out = out;                         // 16*129*129
    float* edit  = out + (size_t)NB * EDSZ;     // 16*129*129
    float* geds  = out + (size_t)2 * NB * EDSZ; // 16

    // Workspace layout (all bf16 operands stored once)
    __bf16* Xbf  = (__bf16*)d_ws;                        // 4097*512
    __bf16* W1bf = Xbf  + (size_t)MROWS * DFEAT;         // 512*512
    __bf16* W2bf = W1bf + (size_t)DFEAT * DFEAT;         // 512*512
    __bf16* H1bf = W2bf + (size_t)DFEAT * DFEAT;         // 4097*512
    __bf16* Ebf  = H1bf + (size_t)MROWS * DFEAT;         // 4097*512
    float*  ss   = (float*)(Ebf + (size_t)MROWS * DFEAT);// 4097
    float*  sums = ss + MROWS + 7;                       // 16

    const int NSIDE = 2048 * DFEAT;                      // elements per side
    const int NW    = DFEAT * DFEAT;

    // One-shot operand conversion (everything L2-resident, ~13 MB total)
    cvt_f32_bf16<<<(NSIDE / 8 + 255) / 256, 256, 0, stream>>>(x_s, Xbf, NSIDE);
    cvt_f32_bf16<<<(NSIDE / 8 + 255) / 256, 256, 0, stream>>>(x_t, Xbf + (size_t)NSIDE, NSIDE);
    cvt_f32_bf16<<<1, 64, 0, stream>>>(virt, Xbf + (size_t)2 * NSIDE, DFEAT);
    cvt_f32_bf16<<<(NW / 8 + 255) / 256, 256, 0, stream>>>(W1, W1bf, NW);
    cvt_f32_bf16<<<(NW / 8 + 255) / 256, 256, 0, stream>>>(W2, W2bf, NW);

    const dim3 gtiles((MROWS + 15) / 16, DFEAT / 16);    // 257 x 32
    gemm_relu_wmma<<<gtiles, 32, 0, stream>>>(Xbf,  W1bf, b1, H1bf, MROWS);
    gemm_relu_wmma<<<gtiles, 32, 0, stream>>>(H1bf, W2bf, b2, Ebf,  MROWS);

    row_sumsq<<<MROWS, 64, 0, stream>>>(Ebf, ss);
    cdist_wmma<<<dim3(9, 9, NB), 32, 0, stream>>>(Ebf, ss, edit);
    batch_sum<<<NB, 256, 0, stream>>>(edit, sums);
    batch_scale<<<dim3((EDSZ + 255) / 256, NB), 256, 0, stream>>>(edit, sums);
    hungarian<<<NB, 256, 0, stream>>>(edit, A_out, geds);
}
